// DRL4VRP_86732569575573
// MI455X (gfx1250) — compile-verified, matrix operand order checked by reading, not run on hardware
//
#include <hip/hip_runtime.h>
#include <math.h>

// ---------------------------------------------------------------------------
// DRL4VRP pointer-network decoder for MI455X (gfx1250, wave32, WMMA bf16)
// B=512, S=200, H=128, SS=DS=2
// ---------------------------------------------------------------------------

#define B_   512
#define S_   200
#define H_   128
#define H3_  384
#define H2_  256

typedef __attribute__((ext_vector_type(16))) __bf16 v16bf;
typedef __attribute__((ext_vector_type(8)))  float  v8f;

__device__ __forceinline__ v8f zero8() {
  v8f c;
#pragma unroll
  for (int i = 0; i < 8; ++i) c[i] = 0.0f;
  return c;
}

__device__ __forceinline__ v8f wmma_bf16(v16bf a, v16bf b, v8f c) {
  return __builtin_amdgcn_wmma_f32_16x16x32_bf16(
      /*neg_a=*/false, a, /*neg_b=*/false, b,
      /*c_mod=*/(short)0, c, /*reuse_a=*/false, /*reuse_b=*/false);
}

// A-fragment (16x32 bf16) from an LDS tile with row stride `rs` elements.
// CDNA5 layout: lane<16 -> K = kb+{0..7, 16..23}; lane>=16 -> K = kb+{8..15, 24..31}
__device__ __forceinline__ v16bf load_a_frag(const __bf16* tile, int rs, int kt) {
  const int lane = threadIdx.x & 31;
  const int m    = lane & 15;
  const int sel  = lane >> 4;
  const __bf16* p = tile + m * rs + kt * 32 + sel * 8;
  v16bf a;
#pragma unroll
  for (int i = 0; i < 8; ++i) { a[i] = p[i]; a[8 + i] = p[16 + i]; }
  return a;
}

// B-fragment from the pre-packed panel: one contiguous 32B load per lane.
__device__ __forceinline__ v16bf load_b_frag(const __bf16* pk, int frag) {
  const int lane = threadIdx.x & 31;
  return *(const v16bf*)(pk + ((size_t)frag * 32 + lane) * 16);
}

__device__ __forceinline__ float sigmoidf_(float x) {
  return 1.0f / (1.0f + expf(-x));
}

__device__ __forceinline__ unsigned wang_hash(unsigned s) {
  s = (s ^ 61u) ^ (s >> 16); s *= 9u; s = s ^ (s >> 4);
  s *= 0x27d4eb2du; s = s ^ (s >> 15); return s;
}

// ---------------------------------------------------------------------------
// Pack a weight matrix W[N][ldw] (using cols colOff..colOff+K) into the WMMA
// B-operand panel layout (frag-major, 32 lanes x 16 bf16 per fragment).
// ---------------------------------------------------------------------------
__global__ void drl_pack_b(const float* __restrict__ W, int ldw, int colOff,
                           int Ktiles, __bf16* __restrict__ out, int total) {
  int tid = blockIdx.x * blockDim.x + threadIdx.x;
  if (tid >= total) return;
  int frag = tid >> 5, lane = tid & 31;
  int nt = frag / Ktiles, kt = frag % Ktiles;
  int n = nt * 16 + (lane & 15);
  int sel = lane >> 4;
  __bf16* o = out + ((size_t)frag * 32 + lane) * 16;
#pragma unroll
  for (int e = 0; e < 16; ++e) {
    int k = kt * 32 + ((e < 8) ? (sel * 8 + e) : (16 + sel * 8 + (e - 8)));
    o[e] = (__bf16)W[(size_t)n * ldw + colOff + k];
  }
}

// ---------------------------------------------------------------------------
// Init decoder state: h <- h0 broadcast, x <- x0 broadcast.
// ---------------------------------------------------------------------------
__global__ void drl_init_state(const float* __restrict__ h0, const float* __restrict__ x0,
                               float* __restrict__ hF, __bf16* __restrict__ hB,
                               float* __restrict__ xbuf) {
  int i = blockIdx.x * blockDim.x + threadIdx.x;
  if (i < B_ * H_) {
    float v = h0[i % H_];
    hF[i] = v;
    hB[i] = (__bf16)v;
  }
  if (i < B_ * 2) xbuf[i] = x0[i % 2];
}

// ---------------------------------------------------------------------------
// Encoder: 1x1 convs; write encCat[b][s][0:128]=static_enc, [128:256]=dynamic_enc
// ---------------------------------------------------------------------------
__global__ void drl_encoder(const float* __restrict__ st, const float* __restrict__ dy,
                            const float* __restrict__ sW, const float* __restrict__ sb,
                            const float* __restrict__ dW, const float* __restrict__ db,
                            __bf16* __restrict__ encCat) {
  int b = blockIdx.x;
  int s0 = blockIdx.y * 16;
  int h = threadIdx.x;  // 128
  float w0 = sW[h * 2], w1 = sW[h * 2 + 1], bs = sb[h];
  float d0 = dW[h * 2], d1 = dW[h * 2 + 1], bd = db[h];
  const float* stb = st + (size_t)b * 2 * S_;
  const float* dyb = dy + (size_t)b * 2 * S_;
#pragma unroll
  for (int i = 0; i < 16; ++i) {
    int s = s0 + i;
    if (s < S_) {
      float se = w0 * stb[s] + w1 * stb[S_ + s] + bs;
      float de = d0 * dyb[s] + d1 * dyb[S_ + s] + bd;
      __bf16* row = encCat + ((size_t)b * S_ + s) * H2_;
      row[h]       = (__bf16)se;
      row[H_ + h]  = (__bf16)de;
    }
  }
}

// ---------------------------------------------------------------------------
// Generic bf16 WMMA GEMM: C[M x 128] = A[M x KT*32] * Bpacked.
// Grid.x = M/16 tiles; block = 128 threads (4 waves), each wave -> 2 N-tiles.
// A-frags re-read from LDS inside the k-loop: no register caching, no spills.
// ---------------------------------------------------------------------------
template <int KT>
__global__ __launch_bounds__(128, 1)
void drl_gemm_bf16(const __bf16* __restrict__ A, int lda,
                   const __bf16* __restrict__ Bp, __bf16* __restrict__ C) {
  __shared__ __bf16 atile[16 * KT * 32];
  const int t = threadIdx.x;
  constexpr int K = KT * 32;
  const size_t mb = (size_t)blockIdx.x * 16;

#pragma unroll
  for (int r = 0; r < (16 * K) / 128; ++r) {
    int idx = r * 128 + t;
    int m = idx / K, k = idx % K;
    atile[idx] = A[(mb + m) * lda + k];
  }
  __syncthreads();

  const int w = t >> 5, lane = t & 31;
  const int nt0 = w * 2;

  v8f c0 = zero8(), c1 = zero8();
#pragma unroll
  for (int kt = 0; kt < KT; ++kt) {
    v16bf a = load_a_frag(atile, K, kt);
    c0 = wmma_bf16(a, load_b_frag(Bp, (nt0 + 0) * KT + kt), c0);
    c1 = wmma_bf16(a, load_b_frag(Bp, (nt0 + 1) * KT + kt), c1);
  }

  const int colbase = (lane & 15);
  const int rsel = (lane >> 4) * 8;
#pragma unroll
  for (int v = 0; v < 8; ++v) {
    C[(mb + rsel + v) * H_ + (nt0 + 0) * 16 + colbase] = (__bf16)c0[v];
    C[(mb + rsel + v) * H_ + (nt0 + 1) * 16 + colbase] = (__bf16)c1[v];
  }
}

// ---------------------------------------------------------------------------
// Per-step GRU cell + attention-query projection, all in one kernel.
// Block = 16 batch rows, 4 waves. gi = emb@Wih^T, gh = h@Whh^T via WMMA,
// gates in f32, then q = h_new @ WaH^T via WMMA.
// ---------------------------------------------------------------------------
__global__ __launch_bounds__(128, 1)
void drl_gru_step(const float* __restrict__ xbuf,
                  float* __restrict__ hF, __bf16* __restrict__ hB,
                  const float* __restrict__ Wemb, const float* __restrict__ bemb,
                  const __bf16* __restrict__ WihP, const __bf16* __restrict__ WhhP,
                  const float* __restrict__ bih, const float* __restrict__ bhh,
                  const __bf16* __restrict__ WaHP, float* __restrict__ q) {
  __shared__ __bf16 emb_lds[16 * H_];
  __shared__ __bf16 h_lds[16 * H_];
  __shared__ __bf16 hn_lds[16 * H_];
  __shared__ float  gi_lds[16 * H3_];
  __shared__ float  gh_lds[16 * H3_];
  __shared__ float  xs[32];

  const int t = threadIdx.x;       // 128
  const int mbase = blockIdx.x * 16;

  if (t < 32) xs[t] = xbuf[mbase * 2 + t];
  for (int i = t; i < 16 * H_; i += 128) h_lds[i] = hB[(size_t)mbase * H_ + i];
  __syncthreads();

  // emb[m][t] = x[m]·Wemb[t] + bemb[t]  (SS=2)
  {
    float w0 = Wemb[t * 2], w1 = Wemb[t * 2 + 1], bb = bemb[t];
#pragma unroll
    for (int m = 0; m < 16; ++m)
      emb_lds[m * H_ + t] = (__bf16)(xs[m * 2] * w0 + xs[m * 2 + 1] * w1 + bb);
  }
  __syncthreads();

  // gi / gh GEMMs: [16x128] x [128x384]; A frags re-read from LDS per k-step.
  {
    const int w = t >> 5, lane = t & 31;
    for (int j = 0; j < 6; ++j) {
      int nt = w * 6 + j;
      v8f cgi = zero8(), cgh = zero8();
#pragma unroll
      for (int kt = 0; kt < 4; ++kt) {
        v16bf ae = load_a_frag(emb_lds, H_, kt);
        v16bf ah = load_a_frag(h_lds,  H_, kt);
        cgi = wmma_bf16(ae, load_b_frag(WihP, nt * 4 + kt), cgi);
        cgh = wmma_bf16(ah, load_b_frag(WhhP, nt * 4 + kt), cgh);
      }
      int col = nt * 16 + (lane & 15);
      int rsel = (lane >> 4) * 8;
#pragma unroll
      for (int v = 0; v < 8; ++v) {
        gi_lds[(rsel + v) * H3_ + col] = cgi[v];
        gh_lds[(rsel + v) * H3_ + col] = cgh[v];
      }
    }
  }
  __syncthreads();

  // GRU gates (torch order r, z, n); h_new = (1-z)*n + z*h
  {
    const int j = t;
    float br = bih[j] + bhh[j];
    float bz = bih[H_ + j] + bhh[H_ + j];
    float bin = bih[2 * H_ + j], bhn = bhh[2 * H_ + j];
#pragma unroll 4
    for (int m = 0; m < 16; ++m) {
      float r = sigmoidf_(gi_lds[m * H3_ + j] + gh_lds[m * H3_ + j] + br);
      float z = sigmoidf_(gi_lds[m * H3_ + H_ + j] + gh_lds[m * H3_ + H_ + j] + bz);
      float n = tanhf(gi_lds[m * H3_ + 2 * H_ + j] + bin +
                      r * (gh_lds[m * H3_ + 2 * H_ + j] + bhn));
      size_t gidx = (size_t)(mbase + m) * H_ + j;
      float hp = hF[gidx];
      float hn = (1.0f - z) * n + z * hp;
      hF[gidx] = hn;
      __bf16 hb = (__bf16)hn;
      hn_lds[m * H_ + j] = hb;
      hB[gidx] = hb;
    }
  }
  __syncthreads();

  // q = h_new @ WaH^T  ([16x128] x [128x128])
  {
    const int w = t >> 5, lane = t & 31;
    const int nt0 = w * 2;
    v8f c0 = zero8(), c1 = zero8();
#pragma unroll
    for (int kt = 0; kt < 4; ++kt) {
      v16bf a = load_a_frag(hn_lds, H_, kt);
      c0 = wmma_bf16(a, load_b_frag(WaHP, (nt0 + 0) * 4 + kt), c0);
      c1 = wmma_bf16(a, load_b_frag(WaHP, (nt0 + 1) * 4 + kt), c1);
    }
    int colbase = (lane & 15);
    int rsel = (lane >> 4) * 8;
#pragma unroll
    for (int v = 0; v < 8; ++v) {
      q[(size_t)(mbase + rsel + v) * H_ + (nt0 + 0) * 16 + colbase] = c0[v];
      q[(size_t)(mbase + rsel + v) * H_ + (nt0 + 1) * 16 + colbase] = c1[v];
    }
  }
}

// ---------------------------------------------------------------------------
// Attention: e[s] = va · tanh(Pa[b,s,:] + q[b,:]); softmax; context = attn·enc
// One block per batch element, 8 waves.
// ---------------------------------------------------------------------------
__global__ __launch_bounds__(256, 1)
void drl_attn(const __bf16* __restrict__ Pa, const __bf16* __restrict__ encCat,
              const float* __restrict__ q, const float* __restrict__ va,
              float* __restrict__ ctx) {
  __shared__ float qs[H_];
  __shared__ float vs[H_];
  __shared__ float e[224];
  __shared__ float red[256];

  const int b = blockIdx.x;
  const int t = threadIdx.x;       // 256
  const int wave = t >> 5, lane = t & 31;

  if (t < H_) { qs[t] = q[(size_t)b * H_ + t]; vs[t] = va[t]; }
  __syncthreads();

  for (int s = wave; s < S_; s += 8) {
    const __bf16* row = Pa + ((size_t)b * S_ + s) * H_;
    __builtin_prefetch(row + 8 * H_, 0, 0);
    float acc = 0.0f;
#pragma unroll
    for (int h = lane; h < H_; h += 32)
      acc += vs[h] * tanhf((float)row[h] + qs[h]);
#pragma unroll
    for (int off = 16; off > 0; off >>= 1) acc += __shfl_xor(acc, off);
    if (lane == 0) e[s] = acc;
  }
  __syncthreads();

  // softmax over S
  red[t] = (t < S_) ? e[t] : -3.4e38f;
  __syncthreads();
  for (int off = 128; off > 0; off >>= 1) {
    if (t < off) red[t] = fmaxf(red[t], red[t + off]);
    __syncthreads();
  }
  float mx = red[0];
  __syncthreads();
  float p = (t < S_) ? expf(e[t] - mx) : 0.0f;
  red[t] = p;
  __syncthreads();
  for (int off = 128; off > 0; off >>= 1) {
    if (t < off) red[t] += red[t + off];
    __syncthreads();
  }
  float inv = 1.0f / red[0];
  if (t < S_) e[t] = p * inv;
  __syncthreads();

  // context[b,h] = sum_s attn[s] * static_enc[b,s,h]
  {
    int h = t & 127, half = t >> 7;
    float acc = 0.0f;
    for (int s = half * 100; s < half * 100 + 100; ++s)
      acc += e[s] * (float)encCat[((size_t)b * S_ + s) * H2_ + h];
    red[t] = acc;
  }
  __syncthreads();
  if (t < H_) ctx[(size_t)b * H_ + t] = red[t] + red[t + H_];
}

// ---------------------------------------------------------------------------
// Pointer head + categorical sample + gather next decoder input.
// ---------------------------------------------------------------------------
__global__ __launch_bounds__(256, 1)
void drl_ptr_sample(const __bf16* __restrict__ Pd, const float* __restrict__ ctx,
                    const float* __restrict__ Wd, const float* __restrict__ vd,
                    const float* __restrict__ st, float* __restrict__ xbuf,
                    float* __restrict__ outIdx, float* __restrict__ outProb,
                    int step) {
  __shared__ float cs[H_];
  __shared__ float q2[H_];
  __shared__ float vs[H_];
  __shared__ float e[224];
  __shared__ float red[256];

  const int b = blockIdx.x;
  const int t = threadIdx.x;       // 256
  const int wave = t >> 5, lane = t & 31;

  if (t < H_) { cs[t] = ctx[(size_t)b * H_ + t]; vs[t] = vd[t]; }
  __syncthreads();

  if (t < H_) {                      // q2 = Wd[:,H:2H] @ context
    const float* wr = Wd + (size_t)t * H2_ + H_;
    float acc = 0.0f;
#pragma unroll 4
    for (int k = 0; k < H_; ++k) acc += wr[k] * cs[k];
    q2[t] = acc;
  }
  __syncthreads();

  for (int s = wave; s < S_; s += 8) {
    const __bf16* row = Pd + ((size_t)b * S_ + s) * H_;
    float acc = 0.0f;
#pragma unroll
    for (int h = lane; h < H_; h += 32)
      acc += vs[h] * tanhf((float)row[h] + q2[h]);
#pragma unroll
    for (int off = 16; off > 0; off >>= 1) acc += __shfl_xor(acc, off);
    if (lane == 0) e[s] = acc;
  }
  __syncthreads();

  red[t] = (t < S_) ? e[t] : -3.4e38f;
  __syncthreads();
  for (int off = 128; off > 0; off >>= 1) {
    if (t < off) red[t] = fmaxf(red[t], red[t + off]);
    __syncthreads();
  }
  float mx = red[0];
  __syncthreads();
  float p = (t < S_) ? expf(e[t] - mx) : 0.0f;
  red[t] = p;
  __syncthreads();
  for (int off = 128; off > 0; off >>= 1) {
    if (t < off) red[t] += red[t + off];
    __syncthreads();
  }
  float inv = 1.0f / red[0];
  if (t < S_) e[t] = p * inv;
  __syncthreads();

  if (t == 0) {
    unsigned seed = wang_hash((unsigned)(step * 9781 + b * 6271) ^ 0x9E3779B9u);
    float u = (float)(seed >> 8) * (1.0f / 16777216.0f);
    float c = 0.0f;
    int idx = S_ - 1;
    for (int s = 0; s < S_; ++s) {
      c += e[s];
      if (u <= c) { idx = s; break; }
    }
    outIdx[(size_t)b * S_ + step]  = (float)idx;
    outProb[(size_t)b * S_ + step] = e[idx];
    xbuf[b * 2 + 0] = st[(size_t)b * 2 * S_ + idx];
    xbuf[b * 2 + 1] = st[(size_t)b * 2 * S_ + S_ + idx];
  }
}

// ---------------------------------------------------------------------------
// Host side
// ---------------------------------------------------------------------------
extern "C" void kernel_launch(void* const* d_in, const int* in_sizes, int n_in,
                              void* d_out, int out_size, void* d_ws, size_t ws_size,
                              hipStream_t stream) {
  const float* st   = (const float*)d_in[0];
  const float* dy   = (const float*)d_in[1];
  const float* sW   = (const float*)d_in[2];
  const float* sb   = (const float*)d_in[3];
  const float* dW   = (const float*)d_in[4];
  const float* db   = (const float*)d_in[5];
  const float* x0   = (const float*)d_in[6];
  const float* h0   = (const float*)d_in[7];
  const float* Wemb = (const float*)d_in[8];
  const float* bemb = (const float*)d_in[9];
  const float* Wih  = (const float*)d_in[10];
  const float* Whh  = (const float*)d_in[11];
  const float* bih  = (const float*)d_in[12];
  const float* bhh  = (const float*)d_in[13];
  const float* Wa   = (const float*)d_in[14];
  const float* va   = (const float*)d_in[15];
  const float* Wd   = (const float*)d_in[16];
  const float* vd   = (const float*)d_in[17];

  float* outIdx  = (float*)d_out;
  float* outProb = (float*)d_out + (size_t)B_ * S_;

  // Workspace carve-up (all offsets 256B aligned)
  char* w = (char*)d_ws;
  size_t off = 0;
  auto alloc = [&](size_t bytes) { void* p = w + off; off += (bytes + 255) & ~(size_t)255; return p; };

  __bf16* encCat = (__bf16*)alloc((size_t)B_ * S_ * H2_ * 2);  // 52.4 MB (L2-resident)
  __bf16* Pa     = (__bf16*)alloc((size_t)B_ * S_ * H_  * 2);  // 26.2 MB
  __bf16* Pd     = (__bf16*)alloc((size_t)B_ * S_ * H_  * 2);  // 26.2 MB
  __bf16* WihP   = (__bf16*)alloc((size_t)H3_ * H_ * 2);
  __bf16* WhhP   = (__bf16*)alloc((size_t)H3_ * H_ * 2);
  __bf16* WaHP   = (__bf16*)alloc((size_t)H_ * H_ * 2);
  __bf16* WaP    = (__bf16*)alloc((size_t)H_ * H2_ * 2);
  __bf16* WdSP   = (__bf16*)alloc((size_t)H_ * H_ * 2);
  float*  hF     = (float*) alloc((size_t)B_ * H_ * 4);
  __bf16* hB     = (__bf16*)alloc((size_t)B_ * H_ * 2);
  float*  q      = (float*) alloc((size_t)B_ * H_ * 4);
  float*  ctx    = (float*) alloc((size_t)B_ * H_ * 4);
  float*  xbuf   = (float*) alloc((size_t)B_ * 2 * 4);
  (void)ws_size; (void)in_sizes; (void)n_in; (void)out_size;

  // --- one-time (per call) preprocessing ---
  drl_pack_b<<<(3072 + 127) / 128, 128, 0, stream>>>(Wih, H_,  0,   4, WihP, 3072);
  drl_pack_b<<<(3072 + 127) / 128, 128, 0, stream>>>(Whh, H_,  0,   4, WhhP, 3072);
  drl_pack_b<<<(1024 + 127) / 128, 128, 0, stream>>>(Wa,  H3_, H2_, 4, WaHP, 1024);
  drl_pack_b<<<(2048 + 127) / 128, 128, 0, stream>>>(Wa,  H3_, 0,   8, WaP,  2048);
  drl_pack_b<<<(1024 + 127) / 128, 128, 0, stream>>>(Wd,  H2_, 0,   4, WdSP, 1024);

  drl_init_state<<<(B_ * H_ + 255) / 256, 256, 0, stream>>>(h0, x0, hF, hB, xbuf);

  drl_encoder<<<dim3(B_, (S_ + 15) / 16), 128, 0, stream>>>(st, dy, sW, sb, dW, db, encCat);

  // Pa = encCat @ Wa[:, :2H]^T  (M = B*S = 102400, K = 256, N = 128)
  drl_gemm_bf16<8><<<(B_ * S_) / 16, 128, 0, stream>>>(encCat, H2_, WaP,  Pa);
  // Pd = static_enc @ Wd[:, :H]^T (K = 128)
  drl_gemm_bf16<4><<<(B_ * S_) / 16, 128, 0, stream>>>(encCat, H2_, WdSP, Pd);

  // --- 200 sequential decode steps ---
  for (int t = 0; t < S_; ++t) {
    drl_gru_step<<<B_ / 16, 128, 0, stream>>>(xbuf, hF, hB, Wemb, bemb,
                                              WihP, WhhP, bih, bhh, WaHP, q);
    drl_attn<<<B_, 256, 0, stream>>>(Pa, encCat, q, va, ctx);
    drl_ptr_sample<<<B_, 256, 0, stream>>>(Pd, ctx, Wd, vd, st, xbuf,
                                           outIdx, outProb, t);
  }
}
